// LexicalEncoder_10608569221426
// MI455X (gfx1250) — matrix operand order repaired and checked by reading, not compiled, not run on hardware
//
#include <hip/hip_runtime.h>
#include <hip/hip_bf16.h>

#define B_N   8192
#define D_DIM 512
#define K_DIM 4096
#define STEPS 16
#define ROWS_PER_WAVE 16
#define WAVES_PER_WG  4
#define ROWS_PER_WG   (ROWS_PER_WAVE * WAVES_PER_WG)   // 64
#define KC            32                               // codebook rows per LDS chunk
#define NCHUNK        (K_DIM / KC)                     // 128

typedef __attribute__((ext_vector_type(16))) __bf16 v16bf;
typedef __attribute__((ext_vector_type(8)))  __bf16 v8bf;
typedef __attribute__((ext_vector_type(8)))  float  v8f;

// ---------------------------------------------------------------------------
// Kernel 1: codebook f32 -> bf16 (workspace; streamed from L2 every step)
// ---------------------------------------------------------------------------
__global__ void cvt_codebook_bf16(const float* __restrict__ src,
                                  __bf16* __restrict__ dst, int n) {
    int i = blockIdx.x * blockDim.x + threadIdx.x;
    int stride = gridDim.x * blockDim.x;
    for (; i < n; i += stride) dst[i] = (__bf16)src[i];
}

// ---------------------------------------------------------------------------
// Async global->LDS stage of one 32KB codebook chunk (2048 x b128 transfers,
// 128 threads x 16 each). ASYNCcnt-tracked; no VGPR round trip.
//   GVS mode: mem = SADDR + VADDR(32b) + IOFFSET ; LDS = VDST + IOFFSET
// ---------------------------------------------------------------------------
__device__ __forceinline__ void stage_chunk_async(const __bf16* cbBase, int kt,
                                                  void* ldsDst, int tid) {
    unsigned ldsAddr = (unsigned)(size_t)ldsDst + (unsigned)tid * 16u;  // low32 = LDS offset
    unsigned gOff    = (unsigned)(kt * D_DIM * (int)sizeof(__bf16)) + (unsigned)tid * 16u;
    #pragma unroll
    for (int j = 0; j < 16; ++j) {
        asm volatile("global_load_async_to_lds_b128 %0, %1, %2 offset:%3"
                     :
                     : "v"(ldsAddr), "v"(gOff), "s"(cbBase), "n"(j * 2048)
                     : "memory");
    }
}

__device__ __forceinline__ void wait_async_zero() {
    asm volatile("s_wait_asynccnt 0x0" ::: "memory");
}

// ---------------------------------------------------------------------------
// Kernel 2: persistent greedy residual pursuit.
//   A matrix = codebook rows (M = codebook idx, contraction = D)
//   B matrix = residual rows (N = batch rows)
// ---------------------------------------------------------------------------
__launch_bounds__(WAVES_PER_WG * 32, 1)
__global__ void pursuit_kernel(const float* __restrict__ targets,
                               const float* __restrict__ codebookF,
                               const __bf16* __restrict__ codebookBf,
                               float* __restrict__ resF,     // ws: [B][D] f32 residual
                               float* __restrict__ outSeq,   // [B][16]
                               float* __restrict__ outMask,  // [B][16]
                               float* __restrict__ outRes)   // [B][D]
{
    __shared__ __bf16 cbTile[2][KC][D_DIM];         // 64 KB, double-buffered
    __shared__ __bf16 resBf[ROWS_PER_WG][D_DIM];    // 64 KB, per-wave-private rows
    __shared__ float  tn2[ROWS_PER_WG];
    __shared__ float  rowSign[ROWS_PER_WG];
    __shared__ int    rowK[ROWS_PER_WG];
    __shared__ int    rowActive[ROWS_PER_WG];

    const int tid  = threadIdx.x;
    const int wave = tid >> 5;
    const int lane = tid & 31;
    const int ln   = lane & 15;     // column / row-in-tile selector
    const int hi   = lane >> 4;     // wave half
    const int waveRowBase = wave * ROWS_PER_WAVE;        // local row base
    const int wgRowBase   = blockIdx.x * ROWS_PER_WG;    // global row base

    float dec = 1.0f;  // decay^(step+1), running product

    for (int step = 0; step < STEPS; ++step) {
        dec *= 0.95f;

        // ---- prepare: residual(f32) -> resBf(LDS bf16), per-row ||r||^2 ----
        for (int r = 0; r < ROWS_PER_WAVE; ++r) {
            const int lrow = waveRowBase + r;
            const int grow = wgRowBase + lrow;
            const float* src = (step == 0) ? (targets + (size_t)grow * D_DIM)
                                           : (resF    + (size_t)grow * D_DIM);
            float ss = 0.f;
            #pragma unroll
            for (int j = 0; j < 4; ++j) {
                const int d = (j * 32 + lane) * 4;
                float4 v = *(const float4*)(src + d);
                ss += v.x * v.x + v.y * v.y + v.z * v.z + v.w * v.w;
                __bf16* bd = &resBf[lrow][d];
                bd[0] = (__bf16)v.x; bd[1] = (__bf16)v.y;
                bd[2] = (__bf16)v.z; bd[3] = (__bf16)v.w;
                if (step == 0) *(float4*)(resF + (size_t)grow * D_DIM + d) = v;
            }
            #pragma unroll
            for (int off = 16; off > 0; off >>= 1) ss += __shfl_xor(ss, off, 32);
            if (lane == 0) {
                if (step == 0) tn2[lrow] = ss;
                rowActive[lrow] = (sqrtf(ss) >= 0.01f) && (sqrtf(tn2[lrow]) >= 1e-8f) ? 1 : 0;
            }
        }

        // ---- scan all K: double-buffered async staging + WMMA + argmax ----
        float bestAbs = -1.0f;
        float bestVal = 0.0f;
        int   bestK   = 0;

        stage_chunk_async(codebookBf, 0, &cbTile[0][0][0], tid);

        for (int c = 0; c < NCHUNK; ++c) {
            const int buf = c & 1;
            wait_async_zero();      // my async loads for chunk c are in LDS
            __syncthreads();        // everyone's loads done; everyone past chunk c-1 compute
            if (c + 1 < NCHUNK)     // overlap: stream chunk c+1 while computing chunk c
                stage_chunk_async(codebookBf, (c + 1) * KC, &cbTile[buf ^ 1][0][0], tid);

            const int kt = c * KC;
            v8f acc0 = {}; v8f acc1 = {};
            #pragma unroll
            for (int dc = 0; dc < D_DIM; dc += 32) {
                union U { v16bf v; v8bf h[2]; };
                // B fragment (residual): lane holds row (base+ln), halves at d = dc + hi*8 and d = dc + hi*8 + 16
                U ub;
                {
                    const v8bf* bp = (const v8bf*)&resBf[waveRowBase + ln][dc + hi * 8];
                    ub.h[0] = bp[0]; ub.h[1] = bp[2];
                }
                // A fragments (codebook): lane holds row (t*16+ln),
                //   halves at d = dc + hi*8 and d = dc + hi*8 + 16
                const int d0 = dc + hi * 8;
                U ua0, ua1;
                ua0.h[0] = *(const v8bf*)&cbTile[buf][ln][d0];
                ua0.h[1] = *(const v8bf*)&cbTile[buf][ln][d0 + 16];
                ua1.h[0] = *(const v8bf*)&cbTile[buf][16 + ln][d0];
                ua1.h[1] = *(const v8bf*)&cbTile[buf][16 + ln][d0 + 16];

                acc0 = __builtin_amdgcn_wmma_f32_16x16x32_bf16(
                           false, ua0.v, false, ub.v, (short)0, acc0, false, false);
                acc1 = __builtin_amdgcn_wmma_f32_16x16x32_bf16(
                           false, ua1.v, false, ub.v, (short)0, acc1, false, false);
            }

            // per-lane argmax; ascending k within a lane -> strict '>' keeps first
            #pragma unroll
            for (int r = 0; r < 8; ++r) {
                float v = acc0[r]; float a = fabsf(v);
                int   k = kt + hi * 8 + r;
                if (a > bestAbs) { bestAbs = a; bestK = k; bestVal = v; }
            }
            #pragma unroll
            for (int r = 0; r < 8; ++r) {
                float v = acc1[r]; float a = fabsf(v);
                int   k = kt + 16 + hi * 8 + r;
                if (a > bestAbs) { bestAbs = a; bestK = k; bestVal = v; }
            }
        }

        // ---- combine the two wave halves (lane n <-> lane n+16), SWAPX16 ----
        {
            const int imm = 0x401F; // xor=0x10, or=0, and=0x1f
            float oA = __int_as_float(__builtin_amdgcn_ds_swizzle(__float_as_int(bestAbs), imm));
            float oV = __int_as_float(__builtin_amdgcn_ds_swizzle(__float_as_int(bestVal), imm));
            int   oK = __builtin_amdgcn_ds_swizzle(bestK, imm);
            if (oA > bestAbs || (oA == bestAbs && oK < bestK)) {
                bestAbs = oA; bestK = oK; bestVal = oV;
            }
        }

        // ---- emit seq/mask for this step, publish per-row update params ----
        if (hi == 0) {
            const int lrow = waveRowBase + ln;
            const int grow = wgRowBase + lrow;
            const int act  = rowActive[lrow];
            const float sgn = (bestVal >= 0.0f) ? 1.0f : -1.0f;
            const int signedIdx = (bestVal >= 0.0f) ? bestK : -(bestK + 1);
            outSeq [grow * STEPS + step] = act ? (float)signedIdx : 0.0f;
            outMask[grow * STEPS + step] = act ? 1.0f : 0.0f;
            rowSign[lrow] = sgn;
            rowK[lrow]    = bestK;
        }

        // ---- rank-1 residual update (full f32 precision) ----
        for (int r = 0; r < ROWS_PER_WAVE; ++r) {
            const int lrow = waveRowBase + r;
            if (!rowActive[lrow]) continue;   // uniform branch (LDS scalar)
            const int grow = wgRowBase + lrow;
            const float s  = rowSign[lrow] * dec;
            float* rp = resF + (size_t)grow * D_DIM;
            const float* cp = codebookF + (size_t)rowK[lrow] * D_DIM;
            #pragma unroll
            for (int j = 0; j < 4; ++j) {
                const int d = (j * 32 + lane) * 4;
                float4 rv = *(float4*)(rp + d);
                float4 cv = *(const float4*)(cp + d);
                rv.x -= s * cv.x; rv.y -= s * cv.y;
                rv.z -= s * cv.z; rv.w -= s * cv.w;
                *(float4*)(rp + d) = rv;
            }
        }
    }

    // ---- write final residual ----
    for (int r = 0; r < ROWS_PER_WAVE; ++r) {
        const int grow = wgRowBase + waveRowBase + r;
        const float* rp = resF + (size_t)grow * D_DIM;
        float* op = outRes + (size_t)grow * D_DIM;
        #pragma unroll
        for (int j = 0; j < 4; ++j) {
            const int d = (j * 32 + lane) * 4;
            *(float4*)(op + d) = *(const float4*)(rp + d);
        }
    }
}

// ---------------------------------------------------------------------------
extern "C" void kernel_launch(void* const* d_in, const int* in_sizes, int n_in,
                              void* d_out, int out_size, void* d_ws, size_t ws_size,
                              hipStream_t stream) {
    (void)in_sizes; (void)n_in; (void)out_size; (void)ws_size;
    const float* targets  = (const float*)d_in[0];   // [B, D]
    const float* codebook = (const float*)d_in[1];   // [K, D]

    // workspace layout: [0, 4MB) codebook bf16 ; [4MB, 20MB) residual f32
    __bf16* cbBf = (__bf16*)d_ws;
    float*  resF = (float*)((char*)d_ws + (size_t)K_DIM * D_DIM * sizeof(__bf16));

    float* outSeq  = (float*)d_out;                       // [B,16]
    float* outMask = outSeq + (size_t)B_N * STEPS;        // [B,16]
    float* outRes  = outMask + (size_t)B_N * STEPS;       // [B,D]

    cvt_codebook_bf16<<<1024, 256, 0, stream>>>(codebook, cbBf, K_DIM * D_DIM);

    pursuit_kernel<<<B_N / ROWS_PER_WG, WAVES_PER_WG * 32, 0, stream>>>(
        targets, codebook, cbBf, resF, outSeq, outMask, outRes);
}
